// EncoderLayer_20186346291741
// MI455X (gfx1250) — matrix operand order repaired and checked by reading, not compiled, not run on hardware
//
#include <hip/hip_runtime.h>
#include <hip/hip_bf16.h>
#include <cstdint>

// ---------------- types ----------------
typedef __attribute__((ext_vector_type(16))) __bf16   v16bf;
typedef __attribute__((ext_vector_type(8)))  float    v8f;
typedef __attribute__((ext_vector_type(4)))  unsigned int u32x4;
typedef __attribute__((ext_vector_type(8)))  int      i32x8;
typedef __attribute__((ext_vector_type(4)))  int      i32x4;

#define BB   32
#define NN   577
#define CC   768
#define HH   12
#define DH   64
#define DFF  3072
#define MM   (BB*NN)          // 18464 rows, = 1154 * 16 exactly
#define NKP  608              // keys padded to 19 strips of 32
#define NQT  37               // ceil(577/16) query tiles
#define KC   256              // B-tile K-chunk staged per TDM transfer

__device__ __forceinline__ unsigned short f2bf(float f) {
    union { float f; unsigned int u; } c; c.f = f;
    unsigned int r = c.u + 0x7FFFu + ((c.u >> 16) & 1u);
    return (unsigned short)(r >> 16);
}

union FragU { u32x4 q[2]; v16bf v; };

// A-fragment (16x32 bf16): lane<16 holds K {0..7,16..23}, lane>=16 {8..15,24..31}
// caller pre-adds (hi ? 8 : 0) elements to base; loads at +0B and +32B.
__device__ __forceinline__ v16bf load_afrag(const unsigned short* base) {
    FragU u;
    u.q[0] = *(const u32x4*)(base);
    u.q[1] = *(const u32x4*)(base + 16);
    return u.v;
}
// B-fragment (32x16 bf16): lane<16 holds K {0..7,16..23}, lane>=16 {8..15,24..31}
// caller pre-adds (hi ? 8 : 0) elements to base; loads at +0B and +32B.
__device__ __forceinline__ v16bf load_bfrag(const unsigned short* base) {
    FragU u;
    u.q[0] = *(const u32x4*)(base);
    u.q[1] = *(const u32x4*)(base + 16);
    return u.v;
}

// ---------------- TDM: DMA a 64-row x KC-col bf16 tile of Wt[N][K] into LDS ----------------
// D# per CDNA5 ISA ch8: group0 = {flags/count, lds_addr, global_addr(57b)+type},
// group1 = {wg_mask/data_size, tensor_dim0/1, tile_dim0/1/2, strides}.
// LDS result layout: 64 rows (cols of W) x KC elements, contiguous per row.
__device__ __forceinline__ void tdm_load_b(const unsigned short* Wt, int K, int Nrows,
                                           int n0, int kc0, unsigned int lds_off) {
    uint64_t ga = (uint64_t)(uintptr_t)Wt + ((uint64_t)(unsigned)n0 * (unsigned)K + (unsigned)kc0) * 2u;
    u32x4 g0;
    g0.x = 1u;                                            // count=1, user mode, no gather
    g0.y = lds_off;                                       // LDS byte address
    g0.z = (unsigned)ga;                                  // global_addr[31:0]
    g0.w = ((unsigned)(ga >> 32) & 0x01FFFFFFu) | (2u << 30); // global_addr[56:32], type=2
    i32x8 g1;
    g1[0] = (int)(1u << 16);                              // wg_mask=0, data_size=1 (2 bytes)
    g1[1] = (int)(((unsigned)K & 0xFFFFu) << 16);         // tensor_dim0[15:0] in [31:16]
    g1[2] = (int)((((unsigned)K >> 16) & 0xFFFFu) |
                  (((unsigned)Nrows & 0xFFFFu) << 16));   // tensor_dim0[31:16] | tensor_dim1[15:0]
    g1[3] = (int)((((unsigned)Nrows >> 16) & 0xFFFFu) |
                  ((unsigned)KC << 16));                  // tensor_dim1[31:16] | tile_dim0
    g1[4] = 64;                                           // tile_dim1=64, tile_dim2=0
    g1[5] = K;                                            // tensor_dim0_stride[31:0]
    g1[6] = 0;                                            // stride hi / dim1_stride lo (unused, 2D)
    g1[7] = 0;
    i32x4 z4 = {0, 0, 0, 0};
#if __clang_major__ >= 23
    i32x8 z8 = {0, 0, 0, 0, 0, 0, 0, 0};
    __builtin_amdgcn_tensor_load_to_lds(g0, g1, z4, z4, z8, 0);
#else
    __builtin_amdgcn_tensor_load_to_lds(g0, g1, z4, z4, 0);
#endif
}

// ---------------- weight transpose + bf16 convert: wt[n*K+k] = w[k*N+n] ----------------
__global__ __launch_bounds__(256) void convT_kernel(const float* __restrict__ w,
                                                    unsigned short* __restrict__ wt,
                                                    int K, int N) {
    int idx = blockIdx.x * 256 + threadIdx.x;
    if (idx >= K * N) return;
    int k = idx / N, n = idx % N;
    wt[(size_t)n * K + k] = f2bf(w[idx]);
}

// ---------------- LayerNorm (fp32 in -> bf16 out), one row per block ----------------
__global__ __launch_bounds__(256) void ln_kernel(const float* __restrict__ x,
                                                 const float* __restrict__ g,
                                                 const float* __restrict__ b,
                                                 unsigned short* __restrict__ out) {
    int row = blockIdx.x;
    const float* xr = x + (size_t)row * CC;
    float s1 = 0.f, s2 = 0.f;
    for (int i = threadIdx.x; i < CC; i += 256) { float v = xr[i]; s1 += v; s2 += v * v; }
    #pragma unroll
    for (int mk = 16; mk >= 1; mk >>= 1) {
        s1 += __shfl_xor(s1, mk, 32);
        s2 += __shfl_xor(s2, mk, 32);
    }
    __shared__ float a1[8], a2[8], smu, srs;
    int wv = threadIdx.x >> 5;
    if ((threadIdx.x & 31) == 0) { a1[wv] = s1; a2[wv] = s2; }
    __syncthreads();
    if (threadIdx.x == 0) {
        float t1 = 0.f, t2 = 0.f;
        for (int i = 0; i < 8; ++i) { t1 += a1[i]; t2 += a2[i]; }
        float mu = t1 * (1.f / CC);
        float var = t2 * (1.f / CC) - mu * mu;
        smu = mu; srs = rsqrtf(var + 1e-5f);
    }
    __syncthreads();
    for (int i = threadIdx.x; i < CC; i += 256)
        out[(size_t)row * CC + i] = f2bf((xr[i] - smu) * srs * g[i] + b[i]);
}

// ---------------- bf16 WMMA GEMM with TDM-staged, double-buffered B tiles ----------------
// D = A[M,K] * Wt[N,K]^T + bias.  8 waves/block; wave = 16 rows x 64 cols.
// EPI 0: scatter to q/k/vt head buffers (q scaled by 0.125)
// EPI 1: fp32 out with fp32 residual add
// EPI 2: exact GELU, bf16 out
struct GemmParams {
    const unsigned short* A;
    const unsigned short* Wt;
    const float* bias;
    int M, N, K;
    const float* resid;
    float* outF;
    unsigned short* outBF;
    unsigned short* q;
    unsigned short* k;
    unsigned short* vt;
};

template<int EPI>
__global__ __launch_bounds__(256) void gemm_wmma(GemmParams p) {
    __shared__ alignas(128) unsigned short sB[2 * 64 * KC];   // 64 KB double buffer

    const int tid = threadIdx.x;
    const int wave = tid >> 5;
    const int lane = tid & 31;
    const int lane16 = lane & 15;
    const int hi = lane >> 4;

    const int mtile0 = blockIdx.y * 8 + wave;
    const bool active = (mtile0 * 16 < p.M);
    const int mtile = active ? mtile0 : 0;      // clamp: inactive waves still hit barriers
    const int n0 = blockIdx.x * 64;

    v8f acc[4];
    v8f z = {0.f, 0.f, 0.f, 0.f, 0.f, 0.f, 0.f, 0.f};
    #pragma unroll
    for (int t = 0; t < 4; ++t) acc[t] = z;

    const unsigned short* Arow = p.A + (size_t)(mtile * 16 + lane16) * p.K + hi * 8;
    const unsigned int ldsbase = (unsigned int)(uintptr_t)(&sB[0]);   // low 32b = LDS offset
    const int chunks = p.K / KC;

    if (tid < 32)                               // wave 0 issues TDM (TENSORcnt is per-wave)
        tdm_load_b(p.Wt, p.K, p.N, n0, 0, ldsbase);

    for (int c = 0; c < chunks; ++c) {
        if ((c + 1) < chunks) {
            if (tid < 32)
                tdm_load_b(p.Wt, p.K, p.N, n0, (c + 1) * KC,
                           ldsbase + (unsigned)((c + 1) & 1) * (64 * KC * 2));
            __builtin_amdgcn_s_wait_tensorcnt(1);   // chunk c landed, c+1 in flight
        } else {
            __builtin_amdgcn_s_wait_tensorcnt(0);
        }
        __syncthreads();

        const unsigned short* bufp = sB + (size_t)(c & 1) * 64 * KC;
        #pragma unroll 4
        for (int kl = 0; kl < KC; kl += 32) {
            v16bf af = load_afrag(Arow + c * KC + kl);
            #pragma unroll
            for (int t = 0; t < 4; ++t) {
                v16bf bfr = load_bfrag(bufp + (size_t)(t * 16 + lane16) * KC + kl + hi * 8);
                acc[t] = __builtin_amdgcn_wmma_f32_16x16x32_bf16(
                    false, af, false, bfr, (short)0, acc[t], false, false);
            }
        }
        __syncthreads();                        // buffer free for chunk c+2
    }

    if (!active) return;

    float bias4[4];
    #pragma unroll
    for (int t = 0; t < 4; ++t) bias4[t] = p.bias[n0 + t * 16 + lane16];

    #pragma unroll
    for (int r = 0; r < 8; ++r) {
        const int m = mtile * 16 + r + hi * 8;
        int bb = 0, ntok = 0;
        if (EPI == 0) { bb = m / NN; ntok = m % NN; }
        #pragma unroll
        for (int t = 0; t < 4; ++t) {
            const int col = n0 + t * 16 + lane16;
            float val = acc[t][r] + bias4[t];
            if constexpr (EPI == 0) {
                const int which = col / CC;
                const int rem = col % CC;
                const int h = rem >> 6;
                const int dh = rem & 63;
                const size_t bh = (size_t)(bb * HH + h);
                if (which == 0)      p.q[(bh * NKP + ntok) * DH + dh] = f2bf(val * 0.125f);
                else if (which == 1) p.k[(bh * NKP + ntok) * DH + dh] = f2bf(val);
                else                 p.vt[(bh * DH + dh) * NKP + ntok] = f2bf(val);
            } else if constexpr (EPI == 1) {
                const size_t idx = (size_t)m * p.N + col;
                p.outF[idx] = val + p.resid[idx];
            } else {
                const size_t idx = (size_t)m * p.N + col;
                float gl = 0.5f * val * (1.0f + erff(val * 0.70710678118654752f));
                p.outBF[idx] = f2bf(gl);
            }
        }
    }
}

// ---------------- flash attention: one wave per (b,h, 16-query tile) ----------------
__global__ __launch_bounds__(32) void attn_kernel(const unsigned short* __restrict__ q,
                                                  const unsigned short* __restrict__ k,
                                                  const unsigned short* __restrict__ vt,
                                                  unsigned short* __restrict__ o) {
    const int lane = threadIdx.x;
    const int lane16 = lane & 15;
    const int hi = lane >> 4;
    const int qt = blockIdx.x % NQT;
    const int bh = blockIdx.x / NQT;

    const unsigned short* qb = q  + (size_t)bh * NKP * DH;
    const unsigned short* kb = k  + (size_t)bh * NKP * DH;
    const unsigned short* vb = vt + (size_t)bh * DH * NKP;

    __shared__ unsigned short pls[16 * 32];

    const int qrow = qt * 16 + lane16;   // < 592 <= NKP, zero-padded
    v16bf qa0 = load_afrag(qb + (size_t)qrow * DH + 0  + hi * 8);
    v16bf qa1 = load_afrag(qb + (size_t)qrow * DH + 32 + hi * 8);

    v8f zero = {0.f, 0.f, 0.f, 0.f, 0.f, 0.f, 0.f, 0.f};
    v8f ofr[4];
    #pragma unroll
    for (int t = 0; t < 4; ++t) ofr[t] = zero;
    float m8[8], l8[8];
    #pragma unroll
    for (int r = 0; r < 8; ++r) { m8[r] = -1e30f; l8[r] = 0.f; }

    for (int s = 0; s < NKP / 32; ++s) {
        const int base = s * 32;
        // ---- S strip = Q (16x64) x K^T (64x32) : two 16x16 N-tiles ----
        v16bf kb00 = load_bfrag(kb + (size_t)(base +      lane16) * DH + 0  + hi * 8);
        v16bf kb01 = load_bfrag(kb + (size_t)(base +      lane16) * DH + 32 + hi * 8);
        v16bf kb10 = load_bfrag(kb + (size_t)(base + 16 + lane16) * DH + 0  + hi * 8);
        v16bf kb11 = load_bfrag(kb + (size_t)(base + 16 + lane16) * DH + 32 + hi * 8);
        v8f s0 = __builtin_amdgcn_wmma_f32_16x16x32_bf16(false, qa0, false, kb00, (short)0, zero, false, false);
        s0     = __builtin_amdgcn_wmma_f32_16x16x32_bf16(false, qa1, false, kb01, (short)0, s0,   false, false);
        v8f s1 = __builtin_amdgcn_wmma_f32_16x16x32_bf16(false, qa0, false, kb10, (short)0, zero, false, false);
        s1     = __builtin_amdgcn_wmma_f32_16x16x32_bf16(false, qa1, false, kb11, (short)0, s1,   false, false);

        // ---- mask padded keys ----
        const bool msk0 = (base +      lane16) >= NN;
        const bool msk1 = (base + 16 + lane16) >= NN;
        float sm0[8], sm1[8];
        #pragma unroll
        for (int r = 0; r < 8; ++r) {
            sm0[r] = msk0 ? -1e30f : s0[r];
            sm1[r] = msk1 ? -1e30f : s1[r];
        }
        // ---- row max across 16-lane halves ----
        float rm[8];
        #pragma unroll
        for (int r = 0; r < 8; ++r) rm[r] = fmaxf(sm0[r], sm1[r]);
        #pragma unroll
        for (int mk = 1; mk < 16; mk <<= 1)
            #pragma unroll
            for (int r = 0; r < 8; ++r)
                rm[r] = fmaxf(rm[r], __shfl_xor(rm[r], mk, 32));

        float al[8], p0[8], p1[8], rs[8];
        #pragma unroll
        for (int r = 0; r < 8; ++r) {
            float mn = fmaxf(m8[r], rm[r]);
            al[r] = __expf(m8[r] - mn);
            m8[r] = mn;
            p0[r] = __expf(sm0[r] - mn);
            p1[r] = __expf(sm1[r] - mn);
            rs[r] = p0[r] + p1[r];
        }
        #pragma unroll
        for (int mk = 1; mk < 16; mk <<= 1)
            #pragma unroll
            for (int r = 0; r < 8; ++r)
                rs[r] += __shfl_xor(rs[r], mk, 32);
        #pragma unroll
        for (int r = 0; r < 8; ++r) l8[r] = l8[r] * al[r] + rs[r];
        #pragma unroll
        for (int t = 0; t < 4; ++t)
            #pragma unroll
            for (int r = 0; r < 8; ++r) ofr[t][r] *= al[r];

        // ---- P (D-layout) -> LDS -> A-fragment (16 rows x 32 keys) ----
        #pragma unroll
        for (int r = 0; r < 8; ++r) {
            const int prow = hi * 8 + r;
            pls[prow * 32 +      lane16] = f2bf(p0[r]);
            pls[prow * 32 + 16 + lane16] = f2bf(p1[r]);
        }
        __syncthreads();
        v16bf pa = load_afrag(pls + lane16 * 32 + hi * 8);
        __syncthreads();

        // ---- O += P (16x32) x V (32x64) ----
        #pragma unroll
        for (int t = 0; t < 4; ++t) {
            v16bf vf = load_bfrag(vb + (size_t)(t * 16 + lane16) * NKP + base + hi * 8);
            ofr[t] = __builtin_amdgcn_wmma_f32_16x16x32_bf16(
                false, pa, false, vf, (short)0, ofr[t], false, false);
        }
    }

    // ---- epilogue: normalize, store to [B,N,C] bf16 ----
    float inv[8];
    #pragma unroll
    for (int r = 0; r < 8; ++r) inv[r] = 1.0f / l8[r];
    const int bb = bh / HH, h = bh % HH;
    #pragma unroll
    for (int r = 0; r < 8; ++r) {
        const int qr = qt * 16 + r + hi * 8;
        if (qr < NN) {
            #pragma unroll
            for (int t = 0; t < 4; ++t) {
                const int d = t * 16 + lane16;
                o[((size_t)(bb * NN + qr)) * CC + h * DH + d] = f2bf(ofr[t][r] * inv[r]);
            }
        }
    }
}

// ---------------- host orchestration ----------------
extern "C" void kernel_launch(void* const* d_in, const int* in_sizes, int n_in,
                              void* d_out, int out_size, void* d_ws, size_t ws_size,
                              hipStream_t stream) {
    const float* x      = (const float*)d_in[0];
    const float* ln1_g  = (const float*)d_in[1];
    const float* ln1_b  = (const float*)d_in[2];
    const float* ln2_g  = (const float*)d_in[3];
    const float* ln2_b  = (const float*)d_in[4];
    const float* qkv_w  = (const float*)d_in[5];
    const float* qkv_b  = (const float*)d_in[6];
    const float* proj_w = (const float*)d_in[7];
    const float* proj_b = (const float*)d_in[8];
    const float* fc1_w  = (const float*)d_in[9];
    const float* fc1_b  = (const float*)d_in[10];
    const float* fc2_w  = (const float*)d_in[11];
    const float* fc2_b  = (const float*)d_in[12];

    uint8_t* wsp = (uint8_t*)d_ws;
    size_t off = 0;
    auto alloc = [&](size_t bytes) -> void* {
        void* p = wsp + off;
        off = (off + bytes + 255) & ~(size_t)255;
        return p;
    };

    const size_t HEADBUF = (size_t)BB * HH * NKP * DH * 2;   // 29.9 MB (256-mult)
    unsigned short* h1   = (unsigned short*)alloc((size_t)MM * CC * 2);
    unsigned short* wqT  = (unsigned short*)alloc((size_t)3 * CC * CC * 2);
    unsigned short* wpT  = (unsigned short*)alloc((size_t)CC * CC * 2);
    unsigned short* wf1T = (unsigned short*)alloc((size_t)DFF * CC * 2);
    unsigned short* wf2T = (unsigned short*)alloc((size_t)CC * DFF * 2);
    unsigned short* qbuf = (unsigned short*)alloc(HEADBUF);
    unsigned short* kbuf = (unsigned short*)alloc(HEADBUF);
    unsigned short* vtbuf= (unsigned short*)alloc(HEADBUF);
    unsigned short* obuf = (unsigned short*)alloc((size_t)MM * CC * 2);
    float*          x2   = (float*)         alloc((size_t)MM * CC * 4);
    unsigned short* mid  = (unsigned short*)alloc((size_t)MM * DFF * 2);
    unsigned short* h2   = h1;   // h1 dead after QKV GEMM

    // zero padded head buffers (q,k,vt are contiguous, each a 256-multiple)
    hipMemsetAsync(qbuf, 0, HEADBUF * 3, stream);

    // weight transpose + bf16 convert
    convT_kernel<<<(CC * 3 * CC + 255) / 256, 256, 0, stream>>>(qkv_w, wqT, CC, 3 * CC);
    convT_kernel<<<(CC * CC + 255) / 256, 256, 0, stream>>>(proj_w, wpT, CC, CC);
    convT_kernel<<<(CC * DFF + 255) / 256, 256, 0, stream>>>(fc1_w, wf1T, CC, DFF);
    convT_kernel<<<(DFF * CC + 255) / 256, 256, 0, stream>>>(fc2_w, wf2T, DFF, CC);

    const dim3 blk(256);
    const int gy = (MM / 16 + 7) / 8;   // 145

    // LN1
    ln_kernel<<<MM, blk, 0, stream>>>(x, ln1_g, ln1_b, h1);

    // QKV GEMM + head scatter
    GemmParams g1 = { h1, wqT, qkv_b, MM, 3 * CC, CC,
                      nullptr, nullptr, nullptr, qbuf, kbuf, vtbuf };
    gemm_wmma<0><<<dim3((3 * CC) / 64, gy), blk, 0, stream>>>(g1);

    // attention
    attn_kernel<<<BB * HH * NQT, 32, 0, stream>>>(qbuf, kbuf, vtbuf, obuf);

    // proj + residual -> x2 (fp32)
    GemmParams g2 = { obuf, wpT, proj_b, MM, CC, CC,
                      x, x2, nullptr, nullptr, nullptr, nullptr };
    gemm_wmma<1><<<dim3(CC / 64, gy), blk, 0, stream>>>(g2);

    // LN2
    ln_kernel<<<MM, blk, 0, stream>>>(x2, ln2_g, ln2_b, h2);

    // FC1 + GELU -> mid (bf16)
    GemmParams g3 = { h2, wf1T, fc1_b, MM, DFF, CC,
                      nullptr, nullptr, mid, nullptr, nullptr, nullptr };
    gemm_wmma<2><<<dim3(DFF / 64, gy), blk, 0, stream>>>(g3);

    // FC2 + residual -> d_out (fp32)
    GemmParams g4 = { mid, wf2T, fc2_b, MM, CC, DFF,
                      x2, (float*)d_out, nullptr, nullptr, nullptr, nullptr };
    gemm_wmma<1><<<dim3(CC / 64, gy), blk, 0, stream>>>(g4);

    (void)in_sizes; (void)n_in; (void)out_size; (void)ws_size;
}
